// BlockToChannelAggregate_64355789963656
// MI455X (gfx1250) — matrix operand order repaired and checked by reading, not compiled, not run on hardware
//
#include <hip/hip_runtime.h>

// Problem constants (match the reference)
constexpr int BATCH = 32;
constexpr int NBLK  = 1024;
constexpr int HID   = 768;
constexpr int HID2  = 384;   // H/2
constexpr int NCH   = 64;

typedef _Float16 v8h  __attribute__((ext_vector_type(8)));
typedef _Float16 v16h __attribute__((ext_vector_type(16)));
typedef float    v8f  __attribute__((ext_vector_type(8)));

// ---------------------------------------------------------------------------
// Kernel 0: W1 (H x H/2, f32, row-major) -> W1t (H/2 x H, f16, n-major)
// so each WMMA B-matrix lane reads two contiguous 16-byte runs of K.
// ---------------------------------------------------------------------------
__global__ void w1_transpose_f16(const float* __restrict__ W1,
                                 _Float16* __restrict__ W1t) {
  int idx = blockIdx.x * blockDim.x + threadIdx.x;   // 0 .. HID2*HID-1
  int n = idx / HID;
  int k = idx % HID;
  W1t[idx] = (_Float16)W1[k * HID2 + n];
}

// ---------------------------------------------------------------------------
// Kernel 1: gate[b,n] = tanh(tokens @ W1 + b1) @ W2 + b2, via WMMA f16.
// One workgroup = 64 rows of the 32768 x 768 token matrix (4 row-tiles of 16).
// 4 waves; wave w owns N-tiles {w, w+4, ...} (24 tiles of 16 cols of W1).
// Each B fragment is reused by 4 WMMAs (one per row-tile) -> 4x less W1
// traffic from L2 and 4x higher WMMA density per loop body.
// ---------------------------------------------------------------------------
constexpr int ROWS = 64;                             // token rows per workgroup

__launch_bounds__(128)
__global__ void gate_kernel(const float* __restrict__ tokens,
                            const _Float16* __restrict__ W1t,
                            const float* __restrict__ b1,
                            const float* __restrict__ W2,
                            const float* __restrict__ b2,
                            float* __restrict__ gate_out) {
  constexpr int ROWSTRIDE = HID + 8;                 // 776 f16 = 1552 B (16B-aligned, bank-spread)
  __shared__ __align__(16) _Float16 tokA[ROWS * ROWSTRIDE];   // ~99 KB
  __shared__ float gPart[4][ROWS];

  const int tid  = threadIdx.x;
  const long row0 = (long)blockIdx.x * ROWS;

  // Stage the 64 x 768 f32 token tile into LDS as f16 (float4 coalesced loads)
  const float4* src = reinterpret_cast<const float4*>(tokens + row0 * HID);
  constexpr int NV4 = ROWS * HID / 4;                // 12288 float4
  for (int i = tid; i < NV4; i += 128) {
    int r  = i / (HID / 4);
    int c4 = i % (HID / 4);
    float4 v = src[r * (HID / 4) + c4];
    _Float16* dst = &tokA[r * ROWSTRIDE + c4 * 4];
    dst[0] = (_Float16)v.x; dst[1] = (_Float16)v.y;
    dst[2] = (_Float16)v.z; dst[3] = (_Float16)v.w;
  }
  __syncthreads();

  const int wave   = tid >> 5;
  const int lane   = tid & 31;
  const int nlocal = lane & 15;                      // row (A) / col (B) within tile
  const int half   = lane >> 4;                      // lane half selects K sub-range

  float gp[4][8] = {};                               // gate partials per row-tile

  for (int nt = wave; nt < HID2 / 16; nt += 4) {
    const int n0 = nt * 16;
    const int n  = n0 + nlocal;
    // B-matrix: lane -> column n, K = {k0..+7, k0+16..+23} per lane half, +8 for lanes 16-31
    const _Float16* wrow = W1t + (size_t)n * HID + half * 8;
    if (n + 64 < HID2)                               // prefetch next owned N-tile row
      __builtin_prefetch(W1t + (size_t)(n + 64) * HID, 0, 0);

    v8f acc[4];
    #pragma unroll
    for (int r = 0; r < 4; ++r)
      acc[r] = (v8f){0.f, 0.f, 0.f, 0.f, 0.f, 0.f, 0.f, 0.f};

    #pragma unroll 2
    for (int kt = 0; kt < HID / 32; ++kt) {
      const int k0 = kt * 32;
      // Load B once, reuse for 4 row-tiles
      v8h blo = *(const v8h*)(wrow + k0);
      v8h bhi = *(const v8h*)(wrow + k0 + 16);
      v16h bm = __builtin_shufflevector(blo, bhi,
               0, 1, 2, 3, 4, 5, 6, 7, 8, 9, 10, 11, 12, 13, 14, 15);
      #pragma unroll
      for (int r = 0; r < 4; ++r) {
        // A-matrix per ISA: lanes 0-15 K = {k0..+7, k0+16..+23}, lanes 16-31 +8
        const _Float16* arow =
            &tokA[(r * 16 + nlocal) * ROWSTRIDE + half * 8];
        v8h alo = *(const v8h*)(arow + k0);
        v8h ahi = *(const v8h*)(arow + k0 + 16);
        v16h a = __builtin_shufflevector(alo, ahi,
                 0, 1, 2, 3, 4, 5, 6, 7, 8, 9, 10, 11, 12, 13, 14, 15);
        acc[r] = __builtin_amdgcn_wmma_f32_16x16x32_f16(
                   /*neg_a=*/false, a, /*neg_b=*/false, bm,
                   /*c_mod=*/(short)0, acc[r],
                   /*reuse_a=*/false, /*reuse_b=*/false);
      }
    }

    // Epilogue: h = tanh(acc + b1[n]); gate partial = sum_n h * W2[n]
    const float b1v = b1[n];
    const float w2v = W2[n];
    #pragma unroll
    for (int r = 0; r < 4; ++r) {
      #pragma unroll
      for (int j = 0; j < 8; ++j) {
        // C layout: VGPR j, lanes 0-15 -> M=j ; lanes 16-31 -> M=j+8 ; N=nlocal
        float p = tanhf(acc[r][j] + b1v) * w2v;
        p += __shfl_xor(p, 1, 16);
        p += __shfl_xor(p, 2, 16);
        p += __shfl_xor(p, 4, 16);
        p += __shfl_xor(p, 8, 16);
        gp[r][j] += p;                               // group-sum over the 16 N values
      }
    }
  }

  // Deterministic cross-wave reduction (no float atomics)
  if (nlocal == 0) {
    #pragma unroll
    for (int r = 0; r < 4; ++r)
      #pragma unroll
      for (int j = 0; j < 8; ++j)
        gPart[wave][r * 16 + half * 8 + j] = gp[r][j];
  }
  __syncthreads();
  if (tid < ROWS) {
    float g = gPart[0][tid] + gPart[1][tid] + gPart[2][tid] + gPart[3][tid];
    gate_out[row0 + tid] = g + b2[0];
  }
}

// ---------------------------------------------------------------------------
// Kernel 2: per-batch masked softmax over blocks per channel (deterministic).
// blockIdx.x = b. Threads 0..63 each own one channel serially.
// ---------------------------------------------------------------------------
__global__ void softmax_kernel(const float* __restrict__ gate,
                               const unsigned char* __restrict__ active,
                               const int* __restrict__ cmap,
                               float* __restrict__ wbuf,
                               float* __restrict__ chan_active_out) {
  __shared__ float mC[NCH];
  __shared__ float sC[NCH];
  const int b = blockIdx.x;
  const int t = threadIdx.x;                         // 256 threads

  if (t < NCH) {
    const int c = t;
    float m = -__builtin_huge_valf();
    for (int n = 0; n < NBLK; ++n)
      if (active[b * NBLK + n] && cmap[n] == c)
        m = fmaxf(m, gate[b * NBLK + n]);
    float s = 0.f;
    int cnt = 0;
    if (m > -__builtin_huge_valf()) {
      for (int n = 0; n < NBLK; ++n)
        if (active[b * NBLK + n] && cmap[n] == c) {
          s += __expf(gate[b * NBLK + n] - m);
          ++cnt;
        }
    }
    mC[c] = m;
    sC[c] = s;
    chan_active_out[b * NCH + c] = (cnt > 0) ? 1.0f : 0.0f;
  }
  __syncthreads();

  for (int n = t; n < NBLK; n += 256) {
    float wv = 0.f;
    if (active[b * NBLK + n]) {
      const int c = cmap[n];
      wv = __expf(gate[b * NBLK + n] - mC[c]) / sC[c];
    }
    wbuf[b * NBLK + n] = wv;
  }
}

// ---------------------------------------------------------------------------
// Kernel 3: channel_tokens[b,c,:] = sum_{n: map[n]==c} w[b,n] * tokens[b,n,:]
// grid (NCH, BATCH); 256 threads x 3 H-elements; each active token row read
// exactly once device-wide (coalesced 3 KB rows). Deterministic order in n.
// ---------------------------------------------------------------------------
__launch_bounds__(256, 4)
__global__ void aggregate_kernel(const float* __restrict__ tokens,
                                 const int* __restrict__ cmap,
                                 const float* __restrict__ wbuf,
                                 float* __restrict__ out_tokens) {
  const int c = blockIdx.x;
  const int b = blockIdx.y;
  const int t = threadIdx.x;

  float a0 = 0.f, a1 = 0.f, a2 = 0.f;
  const float* tb = tokens + (size_t)b * NBLK * HID;

  for (int n = 0; n < NBLK; ++n) {
    if (cmap[n] == c) {                              // uniform (scalar) compare
      const float wv = wbuf[b * NBLK + n];           // uniform load
      if (wv != 0.f) {
        const float* row = tb + (size_t)n * HID;
        __builtin_prefetch(row + t + HID, 0, 0);
        a0 = fmaf(wv, row[t], a0);
        a1 = fmaf(wv, row[t + 256], a1);
        a2 = fmaf(wv, row[t + 512], a2);
      }
    }
  }
  float* orow = out_tokens + ((size_t)b * NCH + c) * HID;
  orow[t] = a0;
  orow[t + 256] = a1;
  orow[t + 512] = a2;
}

// ---------------------------------------------------------------------------
// Launch. d_in order: block_tokens, block_active, block_to_channel_map,
//                     W1, b1, W2, b2
// d_out: [B*C*H] channel_tokens f32, then [B*C] channel_active (1.0/0.0)
// Workspace: gate (128KB) | w (128KB) | W1t f16 (576KB)  -> ~832KB
// ---------------------------------------------------------------------------
extern "C" void kernel_launch(void* const* d_in, const int* in_sizes, int n_in,
                              void* d_out, int out_size, void* d_ws, size_t ws_size,
                              hipStream_t stream) {
  (void)in_sizes; (void)n_in; (void)out_size; (void)ws_size;

  const float*         tokens = (const float*)d_in[0];
  const unsigned char* active = (const unsigned char*)d_in[1];
  const int*           cmap   = (const int*)d_in[2];
  const float*         W1     = (const float*)d_in[3];
  const float*         b1     = (const float*)d_in[4];
  const float*         W2     = (const float*)d_in[5];
  const float*         b2     = (const float*)d_in[6];

  float* out_tokens = (float*)d_out;
  float* out_active = out_tokens + (size_t)BATCH * NCH * HID;

  char* ws = (char*)d_ws;
  float*    gate = (float*)ws;                        // 32768 floats
  float*    wbuf = (float*)(ws + 131072);             // 32768 floats
  _Float16* W1t  = (_Float16*)(ws + 262144);          // 294912 halves

  w1_transpose_f16<<<(HID * HID2) / 256, 256, 0, stream>>>(W1, W1t);
  gate_kernel<<<(BATCH * NBLK) / ROWS, 128, 0, stream>>>(tokens, W1t, b1, W2, b2, gate);
  softmax_kernel<<<BATCH, 256, 0, stream>>>(gate, active, cmap, wbuf, out_active);
  aggregate_kernel<<<dim3(NCH, BATCH), 256, 0, stream>>>(tokens, cmap, wbuf, out_tokens);
}
